// ComplexHead_70102456205994
// MI455X (gfx1250) — matrix operand order repaired and checked by reading, not compiled, not run on hardware
//
#include <hip/hip_runtime.h>
#include <math.h>

// ---------------------------------------------------------------------------
// ComplexHead fused attention for gfx1250 (MI455X, wave32, WMMA bf16)
//   B=8, T=2048, C=1024, H=64
// ---------------------------------------------------------------------------

typedef __attribute__((ext_vector_type(16))) __bf16 v16bf;
typedef __attribute__((ext_vector_type(8)))  float  v8f;

union ABf {                     // one WMMA 16-bit A/B fragment = 8 VGPRs
    v16bf v;
    unsigned int u[8];
};

#define NEG_INF (-1e30f)

__device__ __forceinline__ unsigned short f32_to_bf16(float f) {
    unsigned int u = __float_as_uint(f);
    return (unsigned short)((u + 0x7FFFu + ((u >> 16) & 1u)) >> 16);  // RNE
}

__device__ __forceinline__ unsigned int f32x2_to_bf16x2(float lo, float hi) {
    unsigned int ul = __float_as_uint(lo);
    unsigned int uh = __float_as_uint(hi);
    ul = (ul + 0x7FFFu + ((ul >> 16) & 1u)) >> 16;
    uh = (uh + 0x7FFFu + ((uh >> 16) & 1u)) >> 16;
    return (ul & 0xFFFFu) | (uh << 16);
}

__device__ __forceinline__ v8f wmma_bf16(const ABf& a, const ABf& b, v8f c) {
    // D = A(16x32 bf16) * B(32x16 bf16) + C(16x16 f32)
    return __builtin_amdgcn_wmma_f32_16x16x32_bf16(
        false, a.v, false, b.v, (short)0, c, false, false);
}

// ---------------------------------------------------------------------------
// Kernel 1: build stacked complex-projection weight, bf16, column(K)-major.
//   packedW[n][k], n in [0,384), k in [0,2048)
//   n/64: 0=q_r 1=q_i 2=k_r 3=k_i 4=v_r 5=v_i, h = n%64
//   out_r = x_r@W_r - x_i@W_i ; out_i = x_i@W_r + x_r@W_i
// ---------------------------------------------------------------------------
__global__ __launch_bounds__(256)
void pack_weights(const float* __restrict__ Wk_r, const float* __restrict__ Wk_i,
                  const float* __restrict__ Wq_r, const float* __restrict__ Wq_i,
                  const float* __restrict__ Wv_r, const float* __restrict__ Wv_i,
                  unsigned short* __restrict__ packedW) {
    int idx = blockIdx.x * 256 + threadIdx.x;      // 384*2048 total
    int n = idx / 2048;
    int k = idx - n * 2048;
    int blk = n >> 6;
    int h = n & 63;
    const float* Wa;
    const float* Wb;
    float sgn;
    switch (blk) {
        case 0:  Wa = Wq_r; Wb = Wq_i; sgn = -1.f; break;
        case 1:  Wa = Wq_i; Wb = Wq_r; sgn =  1.f; break;
        case 2:  Wa = Wk_r; Wb = Wk_i; sgn = -1.f; break;
        case 3:  Wa = Wk_i; Wb = Wk_r; sgn =  1.f; break;
        case 4:  Wa = Wv_r; Wb = Wv_i; sgn = -1.f; break;
        default: Wa = Wv_i; Wb = Wv_r; sgn =  1.f; break;
    }
    float v = (k < 1024) ? Wa[k * 64 + h] : sgn * Wb[(k - 1024) * 64 + h];
    packedW[n * 2048 + k] = f32_to_bf16(v);
}

// ---------------------------------------------------------------------------
// Kernel 2: QKV projection GEMM. M=16384, K=2048, N=384.
// One wave owns SIX 16x16 tiles (one per output matrix: tiles nt = blk*4+g),
// so the f32->bf16 A fragment is built once per k-step and feeds 6 WMMAs,
// and every store's target array is a compile-time constant (no divergence).
// q/k stored row-major [m][h]; v stored transposed [b][h][t] for the PV GEMM.
// ---------------------------------------------------------------------------
__global__ __launch_bounds__(128)
void qkv_proj(const float* __restrict__ xr, const float* __restrict__ xi,
              const unsigned short* __restrict__ packedW,
              unsigned short* __restrict__ qr, unsigned short* __restrict__ qi,
              unsigned short* __restrict__ kr, unsigned short* __restrict__ ki,
              unsigned short* __restrict__ vrT, unsigned short* __restrict__ viT) {
    const int wid  = blockIdx.x * 4 + (threadIdx.x >> 5);   // 4096 waves
    const int lane = threadIdx.x & 31;
    const int lrow = lane & 15;
    const int half = lane >> 4;
    const int mt = wid >> 2;           // 0..1023 : 16-row M tile
    const int g  = wid & 3;            // 0..3    : 16-col group within each 64-block
    const int gm = mt * 16 + lrow;     // A row for this lane (= b*T + t)

    // B pointers: column ncol = blk*64 + g*16 + lrow, weights stored K-major
    const unsigned int* bw[6];
#pragma unroll
    for (int blk = 0; blk < 6; ++blk) {
        int ncol = blk * 64 + g * 16 + lrow;
        bw[blk] = (const unsigned int*)(packedW + (size_t)ncol * 2048);
    }

    v8f acc[6] = {};
    for (int kk = 0; kk < 2048; kk += 32) {
        const float* src  = (kk < 1024) ? xr : xi;           // K-tile never straddles
        const float* arow = src + (size_t)gm * 1024 + (kk & 1023);
        ABf a;
#pragma unroll
        for (int j = 0; j < 8; ++j) {
            // A layout: lanes 0-15 K=0..15, lanes 16-31 K=16..31
            int ka = half * 16 + 2 * j;
            a.u[j] = f32x2_to_bf16x2(arow[ka], arow[ka + 1]);
        }
#pragma unroll
        for (int blk = 0; blk < 6; ++blk) {
            ABf b;
#pragma unroll
            for (int j = 0; j < 8; ++j) {
                // B layout: lanes 0-15 K=0..15, lanes 16-31 K=16..31 (pairs/VGPR)
                int kb = kk + half * 16 + 2 * j;
                b.u[j] = bw[blk][kb >> 1];
            }
            acc[blk] = wmma_bf16(a, b, acc[blk]);
        }
    }

    const int h = g * 16 + lrow;       // head-dim column, same for all 6 tiles
    unsigned short* const qk[4] = {qr, qi, kr, ki};
#pragma unroll
    for (int blk = 0; blk < 4; ++blk)
#pragma unroll
        for (int j = 0; j < 8; ++j) {
            int m = mt * 16 + half * 8 + j;          // C layout row
            qk[blk][(size_t)m * 64 + h] = f32_to_bf16(acc[blk][j]);
        }
#pragma unroll
    for (int j = 0; j < 8; ++j) {
        int m  = mt * 16 + half * 8 + j;
        int bb = m >> 11, t = m & 2047;
        vrT[((size_t)bb * 64 + h) * 2048 + t] = f32_to_bf16(acc[4][j]);
        viT[((size_t)bb * 64 + h) * 2048 + t] = f32_to_bf16(acc[5][j]);
    }
}

// ---------------------------------------------------------------------------
// Kernel 3: fused complex flash-attention. One wave = 16 query rows.
// Per 32-key chunk: 16 WMMAs (complex scores), magnitude + causal mask +
// online softmax (half-wave shfl reductions), P transposed C->A layout via
// LDS (s_wait_dscnt), then 8 WMMAs for P@V_r and P@V_i.
// ---------------------------------------------------------------------------
__global__ __launch_bounds__(128)
void attn(const unsigned short* __restrict__ qr, const unsigned short* __restrict__ qi,
          const unsigned short* __restrict__ kr, const unsigned short* __restrict__ ki,
          const unsigned short* __restrict__ vrT, const unsigned short* __restrict__ viT,
          float* __restrict__ out) {
    __shared__ float lds[4][16 * 32];                 // per-wave P staging (2KB each)
    const int warp = threadIdx.x >> 5;
    const int lane = threadIdx.x & 31;
    const int lrow = lane & 15;
    const int half = lane >> 4;
    const int tile = blockIdx.x * 4 + warp;           // 1024 q-tiles
    const int b  = tile >> 7;
    const int mt = tile & 127;
    const int m0 = mt * 16;
    const size_t bbase = (size_t)b * 2048;

    // --- load Q fragments (A layout), plus sign-flipped q_r for imag scores ---
    ABf qra[2], qia[2], nqra[2];
    {
        const unsigned int* qrp = (const unsigned int*)(qr + (bbase + m0 + lrow) * 64);
        const unsigned int* qip = (const unsigned int*)(qi + (bbase + m0 + lrow) * 64);
#pragma unroll
        for (int s = 0; s < 2; ++s)
#pragma unroll
            for (int j = 0; j < 8; ++j) {
                int ka = 32 * s + half * 16 + 2 * j;
                unsigned int uq = qrp[ka >> 1];
                qra[s].u[j]  = uq;
                nqra[s].u[j] = uq ^ 0x80008000u;      // -q_r (bf16 sign bits)
                qia[s].u[j]  = qip[ka >> 1];
            }
    }

    v8f acc_r[4] = {}, acc_i[4] = {};                 // 16 x 64 output accumulators
    float mrow[8], lsum[8];
#pragma unroll
    for (int j = 0; j < 8; ++j) { mrow[j] = NEG_INF; lsum[j] = 0.f; }

    const int nIter = (mt >> 1) + 1;                  // ceil((m0+16)/32)
    for (int it = 0; it < nIter; ++it) {
        const int kb = it * 32;

        if (kb + 32 < 2048) {                         // global_prefetch_b8 next K tile
            __builtin_prefetch(kr + (bbase + kb + 32 + lane) * 64, 0, 1);
            __builtin_prefetch(ki + (bbase + kb + 32 + lane) * 64, 0, 1);
        }

        // ---- complex score tiles: 16x32 over H=64 (two K=32 steps) ----
        v8f sc_r[2], sc_i[2];
#pragma unroll
        for (int n = 0; n < 2; ++n) {
            const int key = kb + n * 16 + lrow;       // B column = key index
            const unsigned int* krp = (const unsigned int*)(kr + (bbase + key) * 64);
            const unsigned int* kip = (const unsigned int*)(ki + (bbase + key) * 64);
            ABf kbr[2], kbi[2];
#pragma unroll
            for (int s = 0; s < 2; ++s)
#pragma unroll
                for (int j = 0; j < 8; ++j) {
                    int kk = 32 * s + half * 16 + 2 * j;   // B layout K index (= h)
                    kbr[s].u[j] = krp[kk >> 1];
                    kbi[s].u[j] = kip[kk >> 1];
                }
            v8f cr = {};                               // q_r.k_r + q_i.k_i
            cr = wmma_bf16(qra[0], kbr[0], cr);
            cr = wmma_bf16(qra[1], kbr[1], cr);
            cr = wmma_bf16(qia[0], kbi[0], cr);
            cr = wmma_bf16(qia[1], kbi[1], cr);
            sc_r[n] = cr;
            v8f ci = {};                               // q_i.k_r - q_r.k_i
            ci = wmma_bf16(qia[0], kbr[0], ci);
            ci = wmma_bf16(qia[1], kbr[1], ci);
            ci = wmma_bf16(nqra[0], kbi[0], ci);
            ci = wmma_bf16(nqra[1], kbi[1], ci);
            sc_i[n] = ci;
        }

        // ---- magnitude + causal mask + online softmax (rows = j + 8*half) ----
        float p[2][8];
#pragma unroll
        for (int j = 0; j < 8; ++j) {
            int trow = m0 + half * 8 + j;
            float rmax = NEG_INF;
#pragma unroll
            for (int n = 0; n < 2; ++n) {
                int key = kb + n * 16 + lrow;
                float re = sc_r[n][j], im = sc_i[n][j];
                float mag = sqrtf(re * re + im * im + 1e-4f) * 0.125f;  // H^-0.5
                mag = (key <= trow) ? mag : NEG_INF;
                p[n][j] = mag;
                rmax = fmaxf(rmax, mag);
            }
            rmax = fmaxf(rmax, __shfl_xor(rmax, 1));   // stays inside 16-lane half
            rmax = fmaxf(rmax, __shfl_xor(rmax, 2));
            rmax = fmaxf(rmax, __shfl_xor(rmax, 4));
            rmax = fmaxf(rmax, __shfl_xor(rmax, 8));
            float mnew  = fmaxf(mrow[j], rmax);
            float scale = __expf(mrow[j] - mnew);
            float rs = 0.f;
#pragma unroll
            for (int n = 0; n < 2; ++n) {
                float e = __expf(p[n][j] - mnew);
                p[n][j] = e;
                rs += e;
            }
            rs += __shfl_xor(rs, 1);
            rs += __shfl_xor(rs, 2);
            rs += __shfl_xor(rs, 4);
            rs += __shfl_xor(rs, 8);
            lsum[j] = lsum[j] * scale + rs;
            mrow[j] = mnew;
#pragma unroll
            for (int t4 = 0; t4 < 4; ++t4) { acc_r[t4][j] *= scale; acc_i[t4][j] *= scale; }
            // stage P (C layout) to LDS: [row][key_local]
            lds[warp][(half * 8 + j) * 32 + lrow]      = p[0][j];
            lds[warp][(half * 8 + j) * 32 + 16 + lrow] = p[1][j];
        }
        asm volatile("s_wait_dscnt 0x0" ::: "memory");  // wave-local LDS RAW fence

        // ---- reload P as a 16x32 bf16 A fragment ----
        ABf pa;
#pragma unroll
        for (int j = 0; j < 8; ++j) {
            int ka = half * 16 + 2 * j;
            pa.u[j] = f32x2_to_bf16x2(lds[warp][lrow * 32 + ka],
                                      lds[warp][lrow * 32 + ka + 1]);
        }
        asm volatile("s_wait_dscnt 0x0" ::: "memory");  // before next-iter overwrite

        // ---- P @ V_r and P @ V_i (K=32 keys, N=64 head dims in 4 tiles) ----
#pragma unroll
        for (int t4 = 0; t4 < 4; ++t4) {
            const int h = t4 * 16 + lrow;               // B column = head dim
            const unsigned int* vrp = (const unsigned int*)(vrT + ((size_t)b * 64 + h) * 2048);
            const unsigned int* vip = (const unsigned int*)(viT + ((size_t)b * 64 + h) * 2048);
            ABf vbr, vbi;
#pragma unroll
            for (int j = 0; j < 8; ++j) {
                int key = kb + half * 16 + 2 * j;        // B layout K index
                vbr.u[j] = vrp[key >> 1];
                vbi.u[j] = vip[key >> 1];
            }
            acc_r[t4] = wmma_bf16(pa, vbr, acc_r[t4]);
            acc_i[t4] = wmma_bf16(pa, vbi, acc_i[t4]);
        }
    }

    // ---- epilogue: normalize and store fp32 (out_real then out_imag) ----
#pragma unroll
    for (int j = 0; j < 8; ++j) {
        float inv = 1.0f / lsum[j];
        int t = m0 + half * 8 + j;
#pragma unroll
        for (int t4 = 0; t4 < 4; ++t4) {
            int h = t4 * 16 + lrow;
            size_t o = ((size_t)b * 2048 + t) * 64 + h;
            out[o]           = acc_r[t4][j] * inv;
            out[o + 1048576] = acc_i[t4][j] * inv;      // + B*T*H floats
        }
    }
}

// ---------------------------------------------------------------------------
extern "C" void kernel_launch(void* const* d_in, const int* in_sizes, int n_in,
                              void* d_out, int out_size, void* d_ws, size_t ws_size,
                              hipStream_t stream) {
    const float* xr   = (const float*)d_in[0];
    const float* xi   = (const float*)d_in[1];
    const float* Wk_r = (const float*)d_in[2];
    const float* Wk_i = (const float*)d_in[3];
    const float* Wq_r = (const float*)d_in[4];
    const float* Wq_i = (const float*)d_in[5];
    const float* Wv_r = (const float*)d_in[6];
    const float* Wv_i = (const float*)d_in[7];

    char* ws = (char*)d_ws;
    unsigned short* packedW = (unsigned short*)(ws + 0x000000); // 1.5 MB
    unsigned short* qr      = (unsigned short*)(ws + 0x180000); // 2 MB each
    unsigned short* qi      = (unsigned short*)(ws + 0x380000);
    unsigned short* kr      = (unsigned short*)(ws + 0x580000);
    unsigned short* ki      = (unsigned short*)(ws + 0x780000);
    unsigned short* vrT     = (unsigned short*)(ws + 0x980000);
    unsigned short* viT     = (unsigned short*)(ws + 0xB80000);
    float* out = (float*)d_out;

    // 1) pack stacked complex weights (bf16, K-major): 384*2048 elements
    pack_weights<<<3072, 256, 0, stream>>>(Wk_r, Wk_i, Wq_r, Wq_i, Wv_r, Wv_i, packedW);
    // 2) QKV projection GEMM: 1024 M-tiles x 4 col-groups, 6 WMMA tiles/wave
    qkv_proj<<<1024, 128, 0, stream>>>(xr, xi, packedW, qr, qi, kr, ki, vrT, viT);
    // 3) fused complex flash-attention: 1024 q-tiles, 4 waves/block
    attn<<<256, 128, 0, stream>>>(qr, qi, kr, ki, vrT, viT, out);
}
